// AdaptiveSlotAttention_45672682225770
// MI455X (gfx1250) — matrix operand order repaired and checked by reading, not compiled, not run on hardware
//
#include <hip/hip_runtime.h>
#include <hip/hip_bf16.h>

// ---------------------------------------------------------------------------
// Types for CDNA5 WMMA (gfx1250, wave32)
// ---------------------------------------------------------------------------
typedef __attribute__((ext_vector_type(16))) __bf16 v16bf;
typedef __attribute__((ext_vector_type(2)))  __bf16 v2bf;
typedef __attribute__((ext_vector_type(8)))  float  v8f;
typedef __attribute__((ext_vector_type(4)))  float    f32x4;
typedef __attribute__((ext_vector_type(4)))  unsigned u32x4;
typedef __attribute__((ext_vector_type(4)))  int      i32x4;

union Frag16 {            // 32 bytes: one bf16 A or B fragment for 16x16x32
    v16bf          v;
    unsigned short u[16];
    unsigned       w[8];
    u32x4          q[2];
};

__device__ __forceinline__ unsigned short f2bf(float f) {
    union { float f; unsigned u; } x; x.f = f;
    unsigned r = x.u + 0x7FFFu + ((x.u >> 16) & 1u);   // round-to-nearest-even
    return (unsigned short)(r >> 16);
}
// packed f32x2 -> bf16x2 (single v_cvt_pk_bf16_f32 when available)
__device__ __forceinline__ unsigned pk2bf(float lo, float hi) {
#if __has_builtin(__builtin_amdgcn_cvt_pk_bf16_f32)
    union { v2bf v; unsigned u; } x;
    x.v = __builtin_amdgcn_cvt_pk_bf16_f32(lo, hi);
    return x.u;
#else
    return (unsigned)f2bf(lo) | ((unsigned)f2bf(hi) << 16);
#endif
}
__device__ __forceinline__ float bf2f(unsigned short h) {
    union { unsigned u; float f; } x; x.u = ((unsigned)h) << 16; return x.f;
}
__device__ __forceinline__ v8f vzero8() {
    v8f z;
#pragma unroll
    for (int i = 0; i < 8; ++i) z[i] = 0.f;
    return z;
}
__device__ __forceinline__ v8f wmma_bf16(v16bf a, v16bf b, v8f c) {
    // D = A(16x32 bf16) x B(32x16 bf16) + C(16x16 f32)
    return __builtin_amdgcn_wmma_f32_16x16x32_bf16(
        /*neg_a=*/false, a, /*neg_b=*/false, b,
        /*c_mod=*/(short)0, c, /*reuse_a=*/false, /*reuse_b=*/false);
}
__device__ __forceinline__ float gelu_exact(float v) {
    return 0.5f * v * (1.0f + erff(v * 0.70710678118654752f));
}

// ---------------------------------------------------------------------------
// CDNA5 async global->LDS copy (ASYNCcnt tracked), with portable fallback
// Builtin signature (probe-verified): (v4i AS1*, v4i AS3*, imm, imm)
// ---------------------------------------------------------------------------
#if defined(__gfx1250__) && __has_builtin(__builtin_amdgcn_global_load_async_to_lds_b128) && \
    __has_builtin(__builtin_amdgcn_s_wait_asynccnt)
#define USE_ASYNC_LDS 1
#else
#define USE_ASYNC_LDS 0
#endif

__device__ __forceinline__ void cp16_g2lds(void* lds, const void* g) {
#if USE_ASYNC_LDS
    __builtin_amdgcn_global_load_async_to_lds_b128(
        (__attribute__((address_space(1))) i32x4*)g,
        (__attribute__((address_space(3))) i32x4*)lds, 0, 0);
#else
    *(u32x4*)lds = *(const u32x4*)g;
#endif
}
#if USE_ASYNC_LDS
#define WAIT_ASYNC(n) __builtin_amdgcn_s_wait_asynccnt(n)
#else
#define WAIT_ASYNC(n) ((void)0)
#endif

// ---------------------------------------------------------------------------
// Problem constants
// ---------------------------------------------------------------------------
#define BB   32
#define NN   1024
#define FF   1152
#define DD   512
#define HH   8
#define SS   12
#define SOBJ 8
#define HD   64

// ---------------------------------------------------------------------------
// Generic tiled WMMA GEMM:  C[M,N] = A[M,K] * B[K,N] (+bias) (+epilogue)
// A,B f32 row-major in global, converted to bf16 (packed cvt) into LDS.
// Block = 128x128 C tile, 256 threads (8 waves), each wave 32x64 (2x4 tiles).
// M,N multiples of 128; K multiple of 32 (true for all uses here).
// EPI: 0=f32 store, 1=gelu->f32, 2=bf16 store, 3=resid add -> f32
// ---------------------------------------------------------------------------
#define LDSTRIDE 40   // bf16 elements per LDS row (32 + pad), keeps 16B align

template <int EPI>
__global__ __launch_bounds__(256)
void gemm_bf16_wmma(const float* __restrict__ A, const float* __restrict__ B,
                    const float* __restrict__ bias, const float* __restrict__ resid,
                    void* __restrict__ Cout, int M, int N, int K) {
    __shared__ __align__(16) unsigned short lsA[128 * LDSTRIDE]; // [row][k]
    __shared__ __align__(16) unsigned short lsB[128 * LDSTRIDE]; // [col][k]

    const int tid  = threadIdx.x;
    const int bm   = blockIdx.y * 128;
    const int bn   = blockIdx.x * 128;
    const int wave = tid >> 5, lane = tid & 31;
    const int wm   = wave >> 1, wn = wave & 1;      // 4x2 wave grid
    const int lrow = lane & 15, lhi = lane >> 4;
    (void)M;

    v8f acc[2][4];
#pragma unroll
    for (int i = 0; i < 2; ++i)
#pragma unroll
        for (int j = 0; j < 4; ++j) acc[i][j] = vzero8();

    for (int k0 = 0; k0 < K; k0 += 32) {
        // ---- stage A tile: 128 rows x 32 k (f32 -> bf16, packed) ----
        {
            const int col4 = (tid & 7) * 4;
#pragma unroll
            for (int p = 0; p < 4; ++p) {
                const int row = (tid >> 3) + p * 32;
                f32x4 av = *(const f32x4*)(A + (size_t)(bm + row) * K + k0 + col4);
                *(unsigned*)&lsA[row * LDSTRIDE + col4]     = pk2bf(av[0], av[1]);
                *(unsigned*)&lsA[row * LDSTRIDE + col4 + 2] = pk2bf(av[2], av[3]);
            }
        }
        // ---- stage B tile transposed: lsB[col][k], k-row pairs packed ----
        {
            const int k = (tid >> 4) * 2;             // 0,2,..,30
#pragma unroll
            for (int p = 0; p < 2; ++p) {
                const int col4 = (tid & 15) * 4 + p * 64;
                f32x4 b0 = *(const f32x4*)(B + (size_t)(k0 + k) * N + bn + col4);
                f32x4 b1 = *(const f32x4*)(B + (size_t)(k0 + k + 1) * N + bn + col4);
#pragma unroll
                for (int i = 0; i < 4; ++i)
                    *(unsigned*)&lsB[(col4 + i) * LDSTRIDE + k] = pk2bf(b0[i], b1[i]);
            }
        }
        __syncthreads();

        // ---- 8 WMMAs per wave ----
        Frag16 af[2], bf[4];
#pragma unroll
        for (int ms = 0; ms < 2; ++ms) {
            const int row = wm * 32 + ms * 16 + lrow;
            const int kb  = lhi * 8;
            af[ms].q[0] = *(const u32x4*)&lsA[row * LDSTRIDE + kb];       // K kb..kb+7
            af[ms].q[1] = *(const u32x4*)&lsA[row * LDSTRIDE + 16 + kb];  // K 16+kb..
        }
#pragma unroll
        for (int ns = 0; ns < 4; ++ns) {
            const int col = wn * 64 + ns * 16 + lrow;
            const int kb  = lhi * 8;
            bf[ns].q[0] = *(const u32x4*)&lsB[col * LDSTRIDE + kb];
            bf[ns].q[1] = *(const u32x4*)&lsB[col * LDSTRIDE + 16 + kb];
        }
#pragma unroll
        for (int ms = 0; ms < 2; ++ms)
#pragma unroll
            for (int ns = 0; ns < 4; ++ns)
                acc[ms][ns] = wmma_bf16(af[ms].v, bf[ns].v, acc[ms][ns]);
        __syncthreads();
    }

    // ---- epilogue ----
#pragma unroll
    for (int ms = 0; ms < 2; ++ms) {
#pragma unroll
        for (int ns = 0; ns < 4; ++ns) {
            const int col = bn + wn * 64 + ns * 16 + lrow;
            const float bv = bias ? bias[col] : 0.f;
#pragma unroll
            for (int r = 0; r < 8; ++r) {
                const int row = bm + wm * 32 + ms * 16 + r + lhi * 8;
                float v = acc[ms][ns][r] + bv;
                if (EPI == 1) v = gelu_exact(v);
                if (EPI == 3) v += resid[(size_t)row * N + col];
                if (EPI == 2) ((unsigned short*)Cout)[(size_t)row * N + col] = f2bf(v);
                else          ((float*)Cout)[(size_t)row * N + col] = v;
            }
        }
    }
}

// ---------------------------------------------------------------------------
// Row LayerNorm (512 cols), 256 threads per row
// ---------------------------------------------------------------------------
__global__ void ln512(const float* __restrict__ in, float* __restrict__ out,
                      const float* __restrict__ g, const float* __restrict__ b) {
    const int row = blockIdx.x, t = threadIdx.x;
    __shared__ float red[256];
    const float* x = in + (size_t)row * DD;
    float a0 = x[t], a1 = x[t + 256];
    red[t] = a0 + a1; __syncthreads();
    for (int s = 128; s > 0; s >>= 1) { if (t < s) red[t] += red[t + s]; __syncthreads(); }
    const float m = red[0] * (1.f / DD); __syncthreads();
    float d0 = a0 - m, d1 = a1 - m;
    red[t] = d0 * d0 + d1 * d1; __syncthreads();
    for (int s = 128; s > 0; s >>= 1) { if (t < s) red[t] += red[t + s]; __syncthreads(); }
    const float inv = rsqrtf(red[0] * (1.f / DD) + 1e-5f);
    out[(size_t)row * DD + t]       = d0 * inv * g[t]       + b[t];
    out[(size_t)row * DD + t + 256] = d1 * inv * g[t + 256] + b[t + 256];
}

// Fused double LN: out = ln(ln(in, g1,b1), g2,b2)   (in-place safe)
__global__ void ln512x2(const float* __restrict__ in, float* __restrict__ out,
                        const float* __restrict__ g1, const float* __restrict__ b1,
                        const float* __restrict__ g2, const float* __restrict__ b2) {
    const int row = blockIdx.x, t = threadIdx.x;
    __shared__ float red[256];
    const float* x = in + (size_t)row * DD;
    float a0 = x[t], a1 = x[t + 256];
    red[t] = a0 + a1; __syncthreads();
    for (int s = 128; s > 0; s >>= 1) { if (t < s) red[t] += red[t + s]; __syncthreads(); }
    float m = red[0] * (1.f / DD); __syncthreads();
    float d0 = a0 - m, d1 = a1 - m;
    red[t] = d0 * d0 + d1 * d1; __syncthreads();
    for (int s = 128; s > 0; s >>= 1) { if (t < s) red[t] += red[t + s]; __syncthreads(); }
    float inv = rsqrtf(red[0] * (1.f / DD) + 1e-5f); __syncthreads();
    float y0 = d0 * inv * g1[t] + b1[t];
    float y1 = d1 * inv * g1[t + 256] + b1[t + 256];
    // second LN
    red[t] = y0 + y1; __syncthreads();
    for (int s = 128; s > 0; s >>= 1) { if (t < s) red[t] += red[t + s]; __syncthreads(); }
    m = red[0] * (1.f / DD); __syncthreads();
    d0 = y0 - m; d1 = y1 - m;
    red[t] = d0 * d0 + d1 * d1; __syncthreads();
    for (int s = 128; s > 0; s >>= 1) { if (t < s) red[t] += red[t + s]; __syncthreads(); }
    inv = rsqrtf(red[0] * (1.f / DD) + 1e-5f);
    out[(size_t)row * DD + t]       = d0 * inv * g2[t]       + b2[t];
    out[(size_t)row * DD + t + 256] = d1 * inv * g2[t + 256] + b2[t + 256];
}

// ---------------------------------------------------------------------------
// slots = slot_mu + exp(slot_log_sigma) * noise
// ---------------------------------------------------------------------------
__global__ void slots_init(const float* __restrict__ mu, const float* __restrict__ lsig,
                           const float* __restrict__ noise, float* __restrict__ slots) {
    const int idx = blockIdx.x * 256 + threadIdx.x;
    if (idx >= BB * SS * DD) return;
    const int msd = idx % (SS * DD);
    slots[idx] = mu[msd] + __expf(lsig[msd]) * noise[idx];
}

// vsum[b,d] = sum_j V[b,j,d]   (V bf16)
__global__ void vsum_kernel(const unsigned short* __restrict__ Vw, float* __restrict__ vsum) {
    const int b = blockIdx.x, t = threadIdx.x;
    for (int d = t; d < DD; d += 256) {
        float s = 0.f;
        for (int j = 0; j < NN; ++j) s += bf2f(Vw[((size_t)b * NN + j) * DD + d]);
        vsum[b * DD + d] = s;
    }
}

// ---------------------------------------------------------------------------
// dots + softmax(axis = slot*head) + fused row-sum atomics
// one block per (b, 128-col j tile).  K tiles double-buffered in LDS via
// async global->LDS copies (ASYNCcnt); q fragments masked branchlessly;
// padded slot rows redirected to an LDS dump row (no exec divergence).
// q f32 (B*S x D), K bf16 (B*N x D); attn f32 out (B, 96, N)
// ---------------------------------------------------------------------------
#define KSTR 72   // bf16 elems per LDS K row (64 + 8 pad): conflict-free b128

__global__ __launch_bounds__(256)
void dots_softmax(const float* __restrict__ qb, const unsigned short* __restrict__ Kw,
                  float* __restrict__ attnw, float* __restrict__ rowsum) {
    const int b = blockIdx.y, jt = blockIdx.x;        // jt 0..7
    const int tid = threadIdx.x;
    const int wave = tid >> 5, lane = tid & 31, lrow = lane & 15, lhi = lane >> 4;
    __shared__ float dl[97 * 132];                    // rows 0..95 live, row 96 = dump
    __shared__ __align__(16) unsigned short lsK[2][128 * KSTR];

    const int jbase = jt * 128;
    const float amask = (lrow < SS) ? 1.f : 0.f;      // zero-pad slot rows 12..15
    const int   srow  = (lrow < SS) ? lrow : 0;       // clamped (branchless)

    // stage head-h K tile (128 j x 64 d bf16) into lsK[buf]
    auto prefetch = [&](int h, int buf) {
#pragma unroll
        for (int p = 0; p < 4; ++p) {
            const int idx = tid + p * 256;            // 0..1023
            const int j = idx >> 3, seg = idx & 7;    // 8 x 16B segments per row
            const unsigned short* g =
                Kw + ((size_t)(b * NN + jbase + j)) * DD + h * HD + seg * 8;
            cp16_g2lds(&lsK[buf][j * KSTR + seg * 8], g);
        }
    };

    prefetch(0, 0);
    for (int h = 0; h < HH; ++h) {
        if (h < HH - 1) { prefetch(h + 1, (h + 1) & 1); WAIT_ASYNC(4); }
        else            { WAIT_ASYNC(0); }
        __syncthreads();                              // tile h visible to all waves

        const unsigned short* lk = &lsK[h & 1][0];
        v8f acc = vzero8();
#pragma unroll
        for (int t = 0; t < 2; ++t) {                 // K=64 contraction = 2 x 32
            Frag16 a, bf;
            const float* qp = qb + ((size_t)(b * SS + srow)) * DD + h * HD + t * 32 + lhi * 8;
            f32x4 q0 = *(const f32x4*)(qp);
            f32x4 q1 = *(const f32x4*)(qp + 4);
            f32x4 q2 = *(const f32x4*)(qp + 16);
            f32x4 q3 = *(const f32x4*)(qp + 20);
#pragma unroll
            for (int i = 0; i < 2; ++i) {
                a.w[i]     = pk2bf(q0[2 * i] * amask, q0[2 * i + 1] * amask);
                a.w[2 + i] = pk2bf(q1[2 * i] * amask, q1[2 * i + 1] * amask);
                a.w[4 + i] = pk2bf(q2[2 * i] * amask, q2[2 * i + 1] * amask);
                a.w[6 + i] = pk2bf(q3[2 * i] * amask, q3[2 * i + 1] * amask);
            }
            const int jl = wave * 16 + lrow;          // this wave's 16 columns
            bf.q[0] = *(const u32x4*)&lk[jl * KSTR + t * 32 + lhi * 8];
            bf.q[1] = *(const u32x4*)&lk[jl * KSTR + t * 32 + lhi * 8 + 16];
            acc = wmma_bf16(a.v, bf.v, acc);
        }
#pragma unroll
        for (int r = 0; r < 8; ++r) {
            const int s = r + lhi * 8;
            const int rowidx = (s < SS) ? (s * HH + h) : 96;  // dump row, no branch
            dl[rowidx * 132 + wave * 16 + lrow] = acc[r] * 0.125f;  // SCALE
        }
        __syncthreads();                              // compute done before buf reuse
    }

    // softmax over the 96 (slot,head) rows, per column
    if (tid < 128) {
        const int c = tid;
        float m = -1e30f;
        for (int r = 0; r < 96; ++r) m = fmaxf(m, dl[r * 132 + c]);
        float sm = 0.f;
        for (int r = 0; r < 96; ++r) { float e = __expf(dl[r * 132 + c] - m); dl[r * 132 + c] = e; sm += e; }
        const float inv = 1.f / sm;
        for (int r = 0; r < 96; ++r) dl[r * 132 + c] *= inv;
    }
    __syncthreads();
    // fused partial row-sums (for attn_n denominator) via global atomics
    if (tid < 96) {
        float s = 0.f;
        for (int c = 0; c < 128; ++c) s += dl[tid * 132 + c];
        atomicAdd(&rowsum[b * 96 + tid], s);
    }
    // write attn tile
    for (int idx = tid; idx < 96 * 128; idx += 256) {
        const int r = idx >> 7, c = idx & 127;
        attnw[((size_t)b * 96 + r) * NN + jt * 128 + c] = dl[r * 132 + c];
    }
}

// ---------------------------------------------------------------------------
// upd[b,s,h*64+d] = (sum_j attn[b,s*8+h,j]*V[b,j,h*64+d] + 1e-8*vsum) / (rowsum+1e-8)
// one block (128 thr = 4 waves) per (h, b); K=1024 contraction, WMMA
// ---------------------------------------------------------------------------
__global__ __launch_bounds__(128)
void upd_kernel(const float* __restrict__ attnw, const unsigned short* __restrict__ Vw,
                const float* __restrict__ vsum, const float* __restrict__ rowsum,
                float* __restrict__ upd) {
    const int h = blockIdx.x, b = blockIdx.y;
    const int tid = threadIdx.x;
    const int wave = tid >> 5, lane = tid & 31, lrow = lane & 15, lhi = lane >> 4;
    const float amask = (lrow < SS) ? 1.f : 0.f;
    const int   srow  = (lrow < SS) ? lrow : 0;
    v8f acc = vzero8();
    const int dcol = h * HD + wave * 16 + lrow;       // output column
    for (int kt = 0; kt < NN / 32; ++kt) {
        Frag16 a, bf;
        const float* ap = attnw + ((size_t)b * 96 + srow * HH + h) * NN + kt * 32 + lhi * 8;
        f32x4 a0 = *(const f32x4*)(ap);
        f32x4 a1 = *(const f32x4*)(ap + 4);
        f32x4 a2 = *(const f32x4*)(ap + 16);
        f32x4 a3 = *(const f32x4*)(ap + 20);
#pragma unroll
        for (int i = 0; i < 2; ++i) {
            a.w[i]     = pk2bf(a0[2 * i] * amask, a0[2 * i + 1] * amask);
            a.w[2 + i] = pk2bf(a1[2 * i] * amask, a1[2 * i + 1] * amask);
            a.w[4 + i] = pk2bf(a2[2 * i] * amask, a2[2 * i + 1] * amask);
            a.w[6 + i] = pk2bf(a3[2 * i] * amask, a3[2 * i + 1] * amask);
        }
        const int jb = kt * 32 + lhi * 8;
#pragma unroll
        for (int i = 0; i < 16; ++i)
            bf.u[i] = Vw[((size_t)b * NN + jb + i + (i & 8)) * DD + dcol];
        acc = wmma_bf16(a.v, bf.v, acc);
    }
#pragma unroll
    for (int r = 0; r < 8; ++r) {
        const int s = r + lhi * 8;
        if (s < SS) {
            const float num = acc[r] + 1e-8f * vsum[b * DD + dcol];
            upd[((size_t)b * SS + s) * DD + dcol] = num / (rowsum[b * 96 + s * HH + h] + 1e-8f);
        }
    }
}

// GRU gates (elementwise), slots updated in place
__global__ void gru_gate(const float* __restrict__ gx, const float* __restrict__ gh,
                         float* __restrict__ slots) {
    const int idx = blockIdx.x * 256 + threadIdx.x;
    if (idx >= BB * SS * DD) return;
    const int row = idx >> 9, c = idx & 511;
    const float xr = gx[(size_t)row * 1536 + c];
    const float xz = gx[(size_t)row * 1536 + 512 + c];
    const float xn = gx[(size_t)row * 1536 + 1024 + c];
    const float hr = gh[(size_t)row * 1536 + c];
    const float hz = gh[(size_t)row * 1536 + 512 + c];
    const float hn = gh[(size_t)row * 1536 + 1024 + c];
    const float hv = slots[idx];
    const float r = 1.f / (1.f + __expf(-(xr + hr)));
    const float z = 1.f / (1.f + __expf(-(xz + hz)));
    const float n = tanhf(xn + r * hn);
    slots[idx] = (1.f - z) * n + z * hv;
}

// selection head: logits[row,2] = relu(obj_row @ w1 + b1) @ w2 + b2
__global__ void sel_kernel(const float* __restrict__ slots, const float* __restrict__ w1,
                           const float* __restrict__ b1, const float* __restrict__ w2,
                           const float* __restrict__ b2, float* __restrict__ logits) {
    const int row = blockIdx.x;                       // 0..255  (b*8+s)
    const int b = row >> 3, s = row & 7;
    const int t = threadIdx.x;                        // 256 = hidden dim
    const float* x = slots + ((size_t)b * SS + s) * DD;
    float accv = b1[t];
    for (int k = 0; k < DD; ++k) accv += x[k] * w1[k * 256 + t];
    const float hdn = fmaxf(accv, 0.f);
    __shared__ float red[512];
    red[t] = hdn * w2[t * 2 + 0];
    red[256 + t] = hdn * w2[t * 2 + 1];
    __syncthreads();
    for (int st = 128; st > 0; st >>= 1) {
        if (t < st) { red[t] += red[t + st]; red[256 + t] += red[256 + t + st]; }
        __syncthreads();
    }
    if (t == 0) { logits[row * 2 + 0] = red[0] + b2[0]; logits[row * 2 + 1] = red[256] + b2[1]; }
}

// decision logic (sequential per batch): argmax==1, then force >=2 active
__global__ void decision_kernel(const float* __restrict__ logits, float* __restrict__ dec_out) {
    const int b = threadIdx.x;
    if (b >= BB) return;
    float dec[SOBJ]; float sum = 0.f;
    for (int s = 0; s < SOBJ; ++s) {
        const float l0 = logits[(b * SOBJ + s) * 2 + 0];
        const float l1 = logits[(b * SOBJ + s) * 2 + 1];
        dec[s] = (l1 > l0) ? 1.f : 0.f;               // argmax(first on ties)==1
        sum += dec[s];
    }
    const float needed = fmaxf(0.f, 2.f - sum);
    float cum = 0.f;
    for (int s = 0; s < SOBJ; ++s) {
        cum += 1.f - dec[s];
        if (dec[s] == 0.f && cum <= needed) dec[s] = 1.f;
        dec_out[b * SOBJ + s] = dec[s];
    }
}

__global__ void obj_kernel(const float* __restrict__ slots, const float* __restrict__ dec,
                           float* __restrict__ obj) {
    const int idx = blockIdx.x * 256 + threadIdx.x;   // 32*8*512
    if (idx >= BB * SOBJ * DD) return;
    const int bs = idx >> 9, d = idx & 511;
    const int b = bs >> 3, s = bs & 7;
    obj[idx] = slots[((size_t)b * SS + s) * DD + d] * dec[bs];
}

__global__ void attn_out_kernel(const float* __restrict__ attnw, float* __restrict__ out) {
    const int idx = blockIdx.x * 256 + threadIdx.x;   // 32*8*1024
    if (idx >= BB * SOBJ * NN) return;
    const int bs = idx >> 10, j = idx & 1023;
    const int b = bs >> 3, s = bs & 7;
    float a = 0.f;
    for (int h = 0; h < HH; ++h) a += attnw[((size_t)b * 96 + s * HH + h) * NN + j];
    out[idx] = a * (1.f / HH);
}

// ---------------------------------------------------------------------------
// Host-side orchestration
// ---------------------------------------------------------------------------
extern "C" void kernel_launch(void* const* d_in, const int* in_sizes, int n_in,
                              void* d_out, int out_size, void* d_ws, size_t ws_size,
                              hipStream_t stream) {
    (void)in_sizes; (void)n_in; (void)out_size; (void)ws_size;
    const float* features = (const float*)d_in[0];
    const float* noise    = (const float*)d_in[1];
    const float* w_in     = (const float*)d_in[2];
    const float* b_in     = (const float*)d_in[3];
    const float* ln1_g    = (const float*)d_in[4];
    const float* ln1_b    = (const float*)d_in[5];
    const float* slot_mu  = (const float*)d_in[6];
    const float* slot_ls  = (const float*)d_in[7];
    const float* wq       = (const float*)d_in[8];
    const float* wk       = (const float*)d_in[9];
    const float* wv       = (const float*)d_in[10];
    const float* gru_wih  = (const float*)d_in[11];
    const float* gru_whh  = (const float*)d_in[12];
    const float* gru_bih  = (const float*)d_in[13];
    const float* gru_bhh  = (const float*)d_in[14];
    const float* nslots_g = (const float*)d_in[15];
    const float* nslots_b = (const float*)d_in[16];
    const float* ninput_g = (const float*)d_in[17];
    const float* ninput_b = (const float*)d_in[18];
    const float* ffn_ln_g = (const float*)d_in[19];
    const float* ffn_ln_b = (const float*)d_in[20];
    const float* ffn_w1   = (const float*)d_in[21];
    const float* ffn_b1   = (const float*)d_in[22];
    const float* ffn_w2   = (const float*)d_in[23];
    const float* ffn_b2   = (const float*)d_in[24];
    const float* sel_w1   = (const float*)d_in[25];
    const float* sel_b1   = (const float*)d_in[26];
    const float* sel_w2   = (const float*)d_in[27];
    const float* sel_b2   = (const float*)d_in[28];

    // ---- workspace layout (bytes) ----
    char* w = (char*)d_ws;
    float*          X     = (float*)w;                         // 32768*512 f32 = 64MB
    unsigned short* Kw    = (unsigned short*)(w + 67108864);   // 32MB bf16
    unsigned short* Vw    = (unsigned short*)(w + 100663296);  // 32MB bf16
    float*          vsum  = (float*)(w + 134217728);           // 64KB
    float*          slots = (float*)(w + 134283264);           // 768KB
    float*          sn    = (float*)(w + 135069696);           // 768KB
    float*          qb    = (float*)(w + 135856128);           // 768KB
    // iteration scratch overlaid on X (free after K/V built):
    float* attn   = X;                                         // 12.6MB
    float* rowsum = (float*)(w + 12582912);
    float* updb   = (float*)(w + 12595200);
    float* gx     = (float*)(w + 13381632);
    float* gh     = (float*)(w + 15740928);
    float* fln    = (float*)(w + 18100224);
    float* tbuf   = (float*)(w + 18886656);
    float* logits = (float*)(w + 22032384);

    float* out_obj  = (float*)d_out;                  // 32*8*512
    float* out_dec  = out_obj + BB * SOBJ * DD;       // 32*8
    float* out_attn = out_dec + BB * SOBJ;            // 32*8*1024

    const int MROWS = BB * NN;                        // 32768
    const int SROWS = BB * SS;                        // 384

    // 0) slots init
    slots_init<<<(BB * SS * DD + 255) / 256, 256, 0, stream>>>(slot_mu, slot_ls, noise, slots);

    // 1) X = features @ w_in + b_in   (32768 x 1152 x 512)
    gemm_bf16_wmma<0><<<dim3(DD / 128, MROWS / 128), 256, 0, stream>>>(
        features, w_in, b_in, nullptr, X, MROWS, DD, FF);
    // 2) inp = ln(ln(X, ln1), ninput)   in-place
    ln512x2<<<MROWS, 256, 0, stream>>>(X, X, ln1_g, ln1_b, ninput_g, ninput_b);
    // 3) K = inp @ wk  -> bf16 ; V = inp @ wv -> bf16
    gemm_bf16_wmma<2><<<dim3(DD / 128, MROWS / 128), 256, 0, stream>>>(
        X, wk, nullptr, nullptr, Kw, MROWS, DD, DD);
    gemm_bf16_wmma<2><<<dim3(DD / 128, MROWS / 128), 256, 0, stream>>>(
        X, wv, nullptr, nullptr, Vw, MROWS, DD, DD);
    // 4) vsum (for attn_n epsilon term)
    vsum_kernel<<<BB, 256, 0, stream>>>(Vw, vsum);

    for (int it = 0; it < 3; ++it) {
        // sn = ln(slots); q = sn @ wq
        ln512<<<SROWS, 256, 0, stream>>>(slots, sn, nslots_g, nslots_b);
        gemm_bf16_wmma<0><<<dim3(DD / 128, SROWS / 128), 256, 0, stream>>>(
            sn, wq, nullptr, nullptr, qb, SROWS, DD, DD);
        // dots + softmax over (slot,head) + fused rowsum
        (void)hipMemsetAsync(rowsum, 0, BB * 96 * sizeof(float), stream);
        dots_softmax<<<dim3(NN / 128, BB), 256, 0, stream>>>(qb, Kw, attn, rowsum);
        // upd = attn_n @ V
        upd_kernel<<<dim3(HH, BB), 128, 0, stream>>>(attn, Vw, vsum, rowsum, updb);
        // GRU
        gemm_bf16_wmma<0><<<dim3(1536 / 128, SROWS / 128), 256, 0, stream>>>(
            updb, gru_wih, gru_bih, nullptr, gx, SROWS, 1536, DD);
        gemm_bf16_wmma<0><<<dim3(1536 / 128, SROWS / 128), 256, 0, stream>>>(
            slots, gru_whh, gru_bhh, nullptr, gh, SROWS, 1536, DD);
        gru_gate<<<(SROWS * DD + 255) / 256, 256, 0, stream>>>(gx, gh, slots);
        // FFN with residual
        ln512<<<SROWS, 256, 0, stream>>>(slots, fln, ffn_ln_g, ffn_ln_b);
        gemm_bf16_wmma<1><<<dim3(2048 / 128, SROWS / 128), 256, 0, stream>>>(
            fln, ffn_w1, ffn_b1, nullptr, tbuf, SROWS, 2048, DD);
        gemm_bf16_wmma<3><<<dim3(DD / 128, SROWS / 128), 256, 0, stream>>>(
            tbuf, ffn_w2, ffn_b2, slots, slots, SROWS, DD, 2048);
    }

    // selection head + decision + outputs
    sel_kernel<<<BB * SOBJ, 256, 0, stream>>>(slots, sel_w1, sel_b1, sel_w2, sel_b2, logits);
    decision_kernel<<<1, 32, 0, stream>>>(logits, out_dec);
    obj_kernel<<<(BB * SOBJ * DD + 255) / 256, 256, 0, stream>>>(slots, out_dec, out_obj);
    attn_out_kernel<<<(BB * SOBJ * NN + 255) / 256, 256, 0, stream>>>(attn, out_attn);
}
